// GNNAnomalyDetector_43284680409626
// MI455X (gfx1250) — compile-verified, matrix operand order checked by reading, not run on hardware
//
#include <hip/hip_runtime.h>

// ---------- WMMA types (CDNA5 / gfx1250, wave32) ----------
typedef __attribute__((ext_vector_type(16))) __bf16 v16bf;
typedef __attribute__((ext_vector_type(8)))  __bf16 v8bf;
typedef __attribute__((ext_vector_type(2)))  __bf16 v2bf;
typedef __attribute__((ext_vector_type(8)))  float  v8f;

#define HD 64          // hidden size H
#define FN 128         // node feature dim
#define FE 32          // edge feature dim

// ---------- helpers ----------
__device__ __forceinline__ unsigned f2ord(float f) {
    unsigned b = __float_as_uint(f);
    return (b & 0x80000000u) ? ~b : (b | 0x80000000u);   // monotone float->uint
}
__device__ __forceinline__ float ord2f(unsigned u) {
    return (u & 0x80000000u) ? __uint_as_float(u & 0x7fffffffu) : __uint_as_float(~u);
}

// ---------- zero scratch ----------
__global__ void k_zero(float* __restrict__ p, long n) {
    long i = (long)blockIdx.x * blockDim.x + threadIdx.x;
    if (i < n) p[i] = 0.f;
}

// ---------- f32 -> bf16 streaming convert (n2 = element pairs) ----------
__global__ void k_cvt_bf16(const float* __restrict__ src, __bf16* __restrict__ dst, long n2) {
    long i = (long)blockIdx.x * blockDim.x + threadIdx.x;
    if (i >= n2) return;
    float2 v = *(const float2*)(src + 2 * i);
    v2bf o; o[0] = (__bf16)v.x; o[1] = (__bf16)v.y;
    *(v2bf*)(dst + 2 * i) = o;
}

// ---------- fold attention projections into small vectors ----------
// v = le_W.T @ ae  (64);  u = enc_W.T @ v (32);  c = enc_b . v
// small: [0..31]=u1 [32]=c1 [33]=sum_ae1 ; [64..95]=u2 [96]=c2 [97]=sum_ae2
__global__ void k_fold(const float* __restrict__ encW, const float* __restrict__ encb,
                       const float* __restrict__ le1W, const float* __restrict__ ae1,
                       const float* __restrict__ le2W, const float* __restrict__ ae2,
                       float* __restrict__ small) {
    __shared__ float v1[HD], v2[HD];
    int t = threadIdx.x;                       // 64 threads
    float s1 = 0.f, s2 = 0.f;
    for (int i = 0; i < HD; ++i) {
        s1 += le1W[i * HD + t] * ae1[i];
        s2 += le2W[i * HD + t] * ae2[i];
    }
    v1[t] = s1; v2[t] = s2;
    __syncthreads();
    if (t < FE) {
        float u1 = 0.f, u2 = 0.f;
        for (int j = 0; j < HD; ++j) {
            u1 += encW[j * FE + t] * v1[j];
            u2 += encW[j * FE + t] * v2[j];
        }
        small[t] = u1; small[64 + t] = u2;
    }
    if (t == 0) {
        float c1 = 0.f, c2 = 0.f;
        for (int j = 0; j < HD; ++j) { c1 += encb[j] * v1[j]; c2 += encb[j] * v2[j]; }
        small[32] = c1; small[96] = c2;
    }
}

// ---------- per-edge attention logits: ae[e] = edge_attr[e].u + c ; also sums for mean ----------
__global__ void k_edge_dot(const float* __restrict__ eattr, float* __restrict__ small,
                           float* __restrict__ ae1o, float* __restrict__ ae2o, int E) {
    int gt = blockIdx.x * blockDim.x + threadIdx.x;
    int e = gt >> 5, lane = gt & 31;
    __shared__ float bs1, bs2;
    if (threadIdx.x == 0) { bs1 = 0.f; bs2 = 0.f; }
    __syncthreads();
    if (e < E) {
        float a  = eattr[(long)e * FE + lane];
        float d1 = a * small[lane];
        float d2 = a * small[64 + lane];
        for (int m = 16; m > 0; m >>= 1) {
            d1 += __shfl_xor(d1, m, 32);
            d2 += __shfl_xor(d2, m, 32);
        }
        if (lane == 0) {
            d1 += small[32]; d2 += small[96];
            ae1o[e] = d1; ae2o[e] = d2;
            atomicAdd(&bs1, d1); atomicAdd(&bs2, d2);
        }
    }
    __syncthreads();
    if (threadIdx.x == 0) { atomicAdd(&small[33], bs1); atomicAdd(&small[97], bs2); }
}

// ---------- WMMA GEMM (pure bf16 operands): C[M,64] = A[M,K] @ W[64,ldw].cols(koff..).T ----------
// Requires M % 16 == 0. blockDim=128 (4 waves). Wave w owns M-tile m0 = 64*blockIdx.x + 16*w
// and all four 16-wide N-tiles: A fragment loaded once per K-step, reused for 4 WMMAs.
// Fragments load directly as 2x b128 bf16 per operand (ISA 7.12.2 layout: lane = row/col,
// hi half-wave takes the k+8 stripe; element j <-> k = (j<8 ? j : j+8) + 8*hi + kk).
// No f32 staging, no conversions in the hot loop -> no spills with full K unroll.
template <int K>
__global__ void k_gemm_wmma(const __bf16* __restrict__ A, const __bf16* __restrict__ W,
                            float* __restrict__ C, int M, int ldw, int koff) {
    int wave = threadIdx.x >> 5;
    int lane = threadIdx.x & 31;
    int hi   = lane >> 4;
    int l15  = lane & 15;
    int m0   = blockIdx.x * 64 + wave * 16;
    if (m0 >= M) return;                        // wave-uniform tile guard (M % 16 == 0)

    const __bf16* arp = A + (long)(m0 + l15) * K;
    __builtin_prefetch(arp, 0, 3);              // warm L2 for this lane's A row

    v8f acc[4] = {};
    #pragma unroll
    for (int kk = 0; kk < K; kk += 32) {
        const __bf16* ap = arp + kk + hi * 8;
        v8bf alo = *(const v8bf*)ap;            // k: +0..7
        v8bf ahi = *(const v8bf*)(ap + 16);     // k: +16..23
        v16bf af = __builtin_shufflevector(alo, ahi, 0, 1, 2, 3, 4, 5, 6, 7,
                                                     8, 9, 10, 11, 12, 13, 14, 15);
        #pragma unroll
        for (int nt = 0; nt < 4; ++nt) {
            const __bf16* wp = W + (long)(nt * 16 + l15) * ldw + koff + kk + hi * 8;
            v8bf blo = *(const v8bf*)wp;
            v8bf bhi = *(const v8bf*)(wp + 16);
            v16bf bfm = __builtin_shufflevector(blo, bhi, 0, 1, 2, 3, 4, 5, 6, 7,
                                                          8, 9, 10, 11, 12, 13, 14, 15);
            acc[nt] = __builtin_amdgcn_wmma_f32_16x16x32_bf16(false, af, false, bfm,
                                                              (short)0, acc[nt], false, false);
        }
    }
    // C/D layout: VGPR r -> row m0 + 8*hi + r, col nt*16 + l15. One base, imm offsets.
    float* cp = C + (long)(m0 + hi * 8) * HD + l15;
    #pragma unroll
    for (int nt = 0; nt < 4; ++nt)
        #pragma unroll
        for (int r = 0; r < 8; ++r)
            cp[r * HD + nt * 16] = acc[nt][r];
}

// ---------- per-node a_src/a_dst dots (one wave per node) ----------
__global__ void k_node_att(const float* __restrict__ h, const float* __restrict__ as,
                           const float* __restrict__ ad, float* __restrict__ a_src,
                           float* __restrict__ a_dst, int N) {
    int gt = blockIdx.x * blockDim.x + threadIdx.x;
    int i = gt >> 5, lane = gt & 31;
    if (i >= N) return;
    float h0 = h[(long)i * HD + lane], h1 = h[(long)i * HD + 32 + lane];
    float s = h0 * as[lane] + h1 * as[32 + lane];
    float d = h0 * ad[lane] + h1 * ad[32 + lane];
    for (int m = 16; m > 0; m >>= 1) {
        s += __shfl_xor(s, m, 32);
        d += __shfl_xor(d, m, 32);
    }
    if (lane == 0) { a_src[i] = s; a_dst[i] = d; }
}

// ---------- pass 1: leaky-relu logits + segment max (edges then self-loops) ----------
__global__ void k_alpha_max(const int* __restrict__ row, const int* __restrict__ col,
                            const float* __restrict__ aeE, const float* __restrict__ a_src,
                            const float* __restrict__ a_dst, const float* __restrict__ small,
                            int sumIdx, float* __restrict__ alpha, unsigned* __restrict__ maxb,
                            int E, int N) {
    int e = blockIdx.x * blockDim.x + threadIdx.x;
    if (e >= E + N) return;
    int r, c; float ae;
    if (e < E) { r = row[e]; c = col[e]; ae = aeE[e]; }
    else       { r = c = e - E; ae = small[sumIdx] / (float)E; }   // self-loop: mean edge logit
    float a = a_src[r] + a_dst[c] + ae;
    a = (a > 0.f) ? a : 0.2f * a;                                  // leaky_relu(0.2)
    alpha[e] = a;
    atomicMax(&maxb[c], f2ord(a));
}

// ---------- pass 2: exp + segment sum ----------
__global__ void k_expsum(const int* __restrict__ col, float* __restrict__ alpha,
                         const unsigned* __restrict__ maxb, float* __restrict__ denom,
                         int E, int N) {
    int e = blockIdx.x * blockDim.x + threadIdx.x;
    if (e >= E + N) return;
    int c = (e < E) ? col[e] : e - E;
    float ex = __expf(alpha[e] - ord2f(maxb[c]));
    alpha[e] = ex;
    atomicAdd(&denom[c], ex);
}

// ---------- pass 3: weighted scatter-add of h[row] into accum[col] (wave per edge) ----------
__global__ void k_aggregate(const int* __restrict__ row, const int* __restrict__ col,
                            const float* __restrict__ h, const float* __restrict__ ex,
                            const float* __restrict__ denom, float* __restrict__ accum,
                            int E, int N) {
    long gt = (long)blockIdx.x * blockDim.x + threadIdx.x;
    long e = gt >> 5; int lane = (int)(gt & 31);
    if (e >= (long)E + N) return;
    int r, c;
    if (e < E) { r = row[e]; c = col[e]; } else { r = c = (int)(e - E); }
    float w = ex[e] / denom[c];
    const float* hr = h + (long)r * HD;
    float*       ac = accum + (long)c * HD;
    atomicAdd(&ac[lane],      hr[lane]      * w);
    atomicAdd(&ac[lane + 32], hr[lane + 32] * w);
}

// ---------- bias (+optional relu) ----------
__global__ void k_finalize(const float* __restrict__ accum, const float* __restrict__ bias,
                           float* __restrict__ out, long n, int do_relu) {
    long i = (long)blockIdx.x * blockDim.x + threadIdx.x;
    if (i >= n) return;
    float v = accum[i] + bias[i & 63];
    if (do_relu) v = (v > 0.f) ? v : 0.f;
    out[i] = v;
}

// ---------- edge scorer: sigmoid( relu(g1[r]+g2[c]+b) . s2 + s2b ) (one wave per edge) ----------
__global__ void k_score(const int* __restrict__ row, const int* __restrict__ col,
                        const float* __restrict__ g1, const float* __restrict__ g2,
                        const float* __restrict__ s1b, const float* __restrict__ s2W,
                        const float* __restrict__ s2b, float* __restrict__ out, int E) {
    int gt = blockIdx.x * blockDim.x + threadIdx.x;
    int e = gt >> 5, lane = gt & 31;
    if (e >= E) return;
    int r = row[e], c = col[e];
    float t0 = g1[(long)r * HD + lane]      + g2[(long)c * HD + lane]      + s1b[lane];
    float t1 = g1[(long)r * HD + 32 + lane] + g2[(long)c * HD + 32 + lane] + s1b[32 + lane];
    t0 = (t0 > 0.f) ? t0 : 0.f;
    t1 = (t1 > 0.f) ? t1 : 0.f;
    float z = t0 * s2W[lane] + t1 * s2W[32 + lane];
    for (int m = 16; m > 0; m >>= 1) z += __shfl_xor(z, m, 32);
    if (lane == 0) out[e] = 1.f / (1.f + __expf(-(z + s2b[0])));
}

// ================== host launcher ==================
extern "C" void kernel_launch(void* const* d_in, const int* in_sizes, int n_in,
                              void* d_out, int out_size, void* d_ws, size_t ws_size,
                              hipStream_t stream) {
    (void)n_in; (void)out_size; (void)ws_size;
    const float* x     = (const float*)d_in[0];
    const float* eat   = (const float*)d_in[1];
    const int*   eidx  = (const int*)  d_in[2];
    const float* encW  = (const float*)d_in[3];
    const float* encb  = (const float*)d_in[4];
    const float* W1    = (const float*)d_in[5];
    const float* le1W  = (const float*)d_in[6];
    const float* as1   = (const float*)d_in[7];
    const float* ad1   = (const float*)d_in[8];
    const float* ae1   = (const float*)d_in[9];
    const float* b1    = (const float*)d_in[10];
    const float* W2    = (const float*)d_in[11];
    const float* le2W  = (const float*)d_in[12];
    const float* as2   = (const float*)d_in[13];
    const float* ad2   = (const float*)d_in[14];
    const float* ae2   = (const float*)d_in[15];
    const float* b2    = (const float*)d_in[16];
    const float* s1W   = (const float*)d_in[17];
    const float* s1b   = (const float*)d_in[18];
    const float* s2W   = (const float*)d_in[19];
    const float* s2b   = (const float*)d_in[20];

    const int N = in_sizes[0] / FN;       // 50000
    const int E = in_sizes[2] / 2;        // 800000
    const int* row = eidx;
    const int* col = eidx + E;
    const long NH = (long)N * HD;

    // workspace layout (floats, then bf16 region; all 16B aligned)
    float* ws     = (float*)d_ws;
    float* small  = ws;                   // 256
    float* hA     = ws + 256;             // N*64  (projected features)
    float* hB     = hA + NH;              // N*64  (layer outputs)
    float* accum  = hB + NH;              // N*64
    float* g1     = accum + NH;           // N*64
    float* g2     = g1 + NH;              // N*64
    float* a_src  = g2 + NH;              // N
    float* a_dst  = a_src + N;            // N
    float* ae1a   = a_dst + N;            // E
    float* ae2a   = ae1a + E;             // E
    float* alpha  = ae2a + E;             // E+N
    unsigned* maxb = (unsigned*)(alpha + E + N);  // N
    float* denom  = (float*)(maxb + N);   // N
    __bf16* xbf   = (__bf16*)(denom + N); // N*128 bf16
    __bf16* hbf   = xbf + (long)N * FN;   // N*64  bf16
    __bf16* w1bf  = hbf + NH;             // 64*128
    __bf16* w2bf  = w1bf + HD * FN;       // 64*64
    __bf16* s1wbf = w2bf + HD * HD;       // 64*128

    auto Z = [&](float* p, long n) {
        k_zero<<<(unsigned)((n + 255) / 256), 256, 0, stream>>>(p, n);
    };
    auto CV = [&](const float* s, __bf16* dst, long n) {      // n even
        k_cvt_bf16<<<(unsigned)((n / 2 + 255) / 256), 256, 0, stream>>>(s, dst, n / 2);
    };
    const int EN = E + N;
    const int gemmGrid = (N + 63) / 64;

    // ---- setup: zero accumulators, fold attention vectors, bf16 conversions ----
    Z(small, 256); Z(accum, NH); Z((float*)maxb, N); Z(denom, N);
    k_fold<<<1, 64, 0, stream>>>(encW, encb, le1W, ae1, le2W, ae2, small);
    k_edge_dot<<<(E * 32 + 255) / 256, 256, 0, stream>>>(eat, small, ae1a, ae2a, E);
    CV(x, xbf, (long)N * FN);
    CV(W1, w1bf, HD * FN);
    CV(W2, w2bf, HD * HD);
    CV(s1W, s1wbf, HD * FN);

    // ---- GAT layer 1 ----
    k_gemm_wmma<FN><<<gemmGrid, 128, 0, stream>>>(xbf, w1bf, hA, N, FN, 0);
    k_node_att<<<(N * 32 + 255) / 256, 256, 0, stream>>>(hA, as1, ad1, a_src, a_dst, N);
    k_alpha_max<<<(EN + 255) / 256, 256, 0, stream>>>(row, col, ae1a, a_src, a_dst, small, 33,
                                                      alpha, maxb, E, N);
    k_expsum<<<(EN + 255) / 256, 256, 0, stream>>>(col, alpha, maxb, denom, E, N);
    k_aggregate<<<(unsigned)(((long)EN * 32 + 255) / 256), 256, 0, stream>>>(row, col, hA, alpha,
                                                                             denom, accum, E, N);
    k_finalize<<<(unsigned)((NH + 255) / 256), 256, 0, stream>>>(accum, b1, hB, NH, 1);

    // ---- GAT layer 2 ----
    Z(accum, NH); Z((float*)maxb, N); Z(denom, N);
    CV(hB, hbf, NH);
    k_gemm_wmma<HD><<<gemmGrid, 128, 0, stream>>>(hbf, w2bf, hA, N, HD, 0);
    k_node_att<<<(N * 32 + 255) / 256, 256, 0, stream>>>(hA, as2, ad2, a_src, a_dst, N);
    k_alpha_max<<<(EN + 255) / 256, 256, 0, stream>>>(row, col, ae2a, a_src, a_dst, small, 97,
                                                      alpha, maxb, E, N);
    k_expsum<<<(EN + 255) / 256, 256, 0, stream>>>(col, alpha, maxb, denom, E, N);
    k_aggregate<<<(unsigned)(((long)EN * 32 + 255) / 256), 256, 0, stream>>>(row, col, hA, alpha,
                                                                             denom, accum, E, N);
    k_finalize<<<(unsigned)((NH + 255) / 256), 256, 0, stream>>>(accum, b2, hB, NH, 0);

    // ---- edge scorer: h converted once, consumed by both halves of s1_W ----
    CV(hB, hbf, NH);
    k_gemm_wmma<HD><<<gemmGrid, 128, 0, stream>>>(hbf, s1wbf, g1, N, 2 * HD, 0);
    k_gemm_wmma<HD><<<gemmGrid, 128, 0, stream>>>(hbf, s1wbf, g2, N, 2 * HD, HD);
    k_score<<<(E * 32 + 255) / 256, 256, 0, stream>>>(row, col, g1, g2, s1b, s2W, s2b,
                                                      (float*)d_out, E);
}
